// Net_309237645443
// MI455X (gfx1250) — matrix operand order repaired and checked by reading, not compiled, not run on hardware
//
#include <hip/hip_runtime.h>

typedef __bf16 bf16;
typedef __attribute__((ext_vector_type(16))) __bf16 v16bf;
typedef __attribute__((ext_vector_type(8)))  float  v8f;

#define N_NODES  16384
#define N_EDGES  131072
#define N_GRAPHS 64
#define N_TOT    (N_EDGES + N_NODES)   // edges + self loops

// ---------------------------------------------------------------- utilities
__global__ void k_zero(float* __restrict__ p, unsigned n) {
  unsigned i = blockIdx.x * 256u + threadIdx.x;
  if (i < n) p[i] = 0.0f;
}

__global__ void k_init_deg(float* __restrict__ deg) {
  unsigned i = blockIdx.x * 256u + threadIdx.x;
  if (i < N_NODES) deg[i] = 1.0f;  // self-loop contributes 1
}

__global__ void k_count_deg(const int* __restrict__ dst, float* __restrict__ deg) {
  unsigned e = blockIdx.x * 256u + threadIdx.x;
  if (e < N_EDGES) atomicAdd(&deg[dst[e]], 1.0f);
}

__global__ void k_dinv(const float* __restrict__ deg, float* __restrict__ dinv) {
  unsigned i = blockIdx.x * 256u + threadIdx.x;
  if (i < N_NODES) dinv[i] = rsqrtf(fmaxf(deg[i], 1.0f));
}

__global__ void k_norm(const int* __restrict__ src, const int* __restrict__ dst,
                       const float* __restrict__ dinv, float* __restrict__ nrm) {
  unsigned e = blockIdx.x * 256u + threadIdx.x;
  if (e >= N_TOT) return;
  if (e < N_EDGES) {
    nrm[e] = dinv[src[e]] * dinv[dst[e]];
  } else {
    int i = e - N_EDGES;
    nrm[e] = dinv[i] * dinv[i];
  }
}

// f32 activations [N_NODES x K] -> bf16 [N_NODES x Kp], zero-padded columns
__global__ void k_act_cast(const float* __restrict__ in, bf16* __restrict__ out,
                           int K, int Kp) {
  unsigned i = blockIdx.x * 256u + threadIdx.x;
  unsigned total = (unsigned)N_NODES * (unsigned)Kp;
  if (i >= total) return;
  unsigned r = i / (unsigned)Kp, c = i % (unsigned)Kp;
  out[i] = (c < (unsigned)K) ? (bf16)in[(size_t)r * K + c] : (bf16)0.0f;
}

// f32 weights W[K x F] -> bf16 transposed Bt[F x Kp], zero-padded K rows.
// LDS-tiled so both global read and write are coalesced.
__global__ void k_wt_cast_t(const float* __restrict__ W, bf16* __restrict__ out,
                            int K, int Kp, int F) {
  __shared__ float tile[32][33];
  const int c0 = blockIdx.x * 32;  // F dimension
  const int r0 = blockIdx.y * 32;  // Kp dimension
  for (int j = threadIdx.y; j < 32; j += 8) {
    int r = r0 + j, c = c0 + threadIdx.x;
    tile[j][threadIdx.x] = (r < K) ? W[(size_t)r * F + c] : 0.0f;
  }
  __syncthreads();
  for (int j = threadIdx.y; j < 32; j += 8) {
    int c = c0 + j, r = r0 + threadIdx.x;
    out[(size_t)c * Kp + r] = (bf16)tile[threadIdx.x][j];
  }
}

// ---------------------------------------------------------------- WMMA GEMM
// C[M,Nn] = A[M,Kp](bf16, row-major) * Bt[Nn,Kp](bf16, N-major = B transposed),
// f32 accumulate. Block = 256 threads = 8 wave32s; block tile 64x64.
// Each wave owns two 16x16 output tiles sharing one A fragment.
// All fragment loads are contiguous global_load_b128 (no LDS, no transpose):
//   A frag : lane(g,mn) reads row mn, K = 16g..16g+15 (contiguous)
//   B frag : lane(g,mn) reads col mn, K = 16g..16g+15 (contiguous in Bt)
__global__ __launch_bounds__(256) void k_gemm(const bf16* __restrict__ A,
                                              const bf16* __restrict__ Bt,
                                              float* __restrict__ C,
                                              int Nn, int Kp) {
  const int tid = threadIdx.x;
  const int w  = tid >> 5;            // wave 0..7
  const int wm = w & 3;               // wave row tile 0..3
  const int wn = w >> 2;              // wave col pair 0..1
  const int l  = tid & 31;            // lane
  const int g  = l >> 4;              // lane half
  const int mn = l & 15;              // row/col within 16

  const int row  = blockIdx.y * 64 + wm * 16 + mn;
  const int col0 = blockIdx.x * 64 + wn * 32 + mn;
  const int col1 = col0 + 16;

  const bf16* pa  = A  + (size_t)row  * Kp + 16 * g;
  const bf16* pb0 = Bt + (size_t)col0 * Kp + 16 * g;
  const bf16* pb1 = Bt + (size_t)col1 * Kp + 16 * g;

  typedef union { v16bf v; uint4 q[2]; } Frag;

  v8f acc0 = {}, acc1 = {};
#pragma unroll 2
  for (int kk = 0; kk < Kp; kk += 32) {
    Frag a, b0, b1;
    a.q[0]  = *(const uint4*)(pa + kk);        // K = kk+16g  .. +7
    a.q[1]  = *(const uint4*)(pa + kk + 8);    // K = kk+16g+8.. +15
    b0.q[0] = *(const uint4*)(pb0 + kk);       // K = kk+16g  .. +7
    b0.q[1] = *(const uint4*)(pb0 + kk + 8);   // K = kk+16g+8.. +15
    b1.q[0] = *(const uint4*)(pb1 + kk);
    b1.q[1] = *(const uint4*)(pb1 + kk + 8);
    acc0 = __builtin_amdgcn_wmma_f32_16x16x32_bf16(false, a.v, false, b0.v,
                                                   (short)0, acc0, false, false);
    acc1 = __builtin_amdgcn_wmma_f32_16x16x32_bf16(false, a.v, false, b1.v,
                                                   (short)0, acc1, false, false);
  }

  // C/D layout: VGPR r -> M = r + 8*g, N = lane%16
#pragma unroll
  for (int r = 0; r < 8; ++r) {
    int grow = blockIdx.y * 64 + wm * 16 + 8 * g + r;
    C[(size_t)grow * Nn + col0] = acc0[r];
    C[(size_t)grow * Nn + col1] = acc1[r];
  }
}

// ---------------------------------------------------------------- scatter-add
// agg[dst] += t[src] * norm, per edge, feature-chunked. gridDim = (N_TOT, F/256)
__global__ void k_scatter(const float* __restrict__ t, float* __restrict__ agg,
                          const int* __restrict__ src, const int* __restrict__ dst,
                          const float* __restrict__ nrm, int F) {
  int e = blockIdx.x;
  int f = blockIdx.y * 256 + threadIdx.x;
  int s, d;
  if (e < N_EDGES) { s = src[e]; d = dst[e]; }
  else             { s = d = e - N_EDGES; }
  float w = nrm[e];
  atomicAdd(&agg[(size_t)d * F + f], t[(size_t)s * F + f] * w);
}

// ---------------------------------------------------------------- bias+relu+stats
// gridDim = (F/256, N_NODES/256). Each thread owns one feature column over 256 rows.
__global__ void k_bias_relu_stats(float* __restrict__ h, const float* __restrict__ bias,
                                  float* __restrict__ ssum, float* __restrict__ ssq, int F) {
  int f = blockIdx.x * 256 + threadIdx.x;
  float b = bias[f];
  float s = 0.0f, s2 = 0.0f;
  int r0 = blockIdx.y * 256;
  for (int r = r0; r < r0 + 256; ++r) {
    size_t idx = (size_t)r * F + f;
    float v = fmaxf(h[idx] + b, 0.0f);
    h[idx] = v;
    s += v; s2 += v * v;
  }
  atomicAdd(&ssum[f], s);
  atomicAdd(&ssq[f], s2);
}

__global__ void k_finalize(const float* __restrict__ ssum, const float* __restrict__ ssq,
                           float* __restrict__ mean, float* __restrict__ istd, int F) {
  int f = blockIdx.x * 256 + threadIdx.x;
  if (f >= F) return;
  const float invn = 1.0f / (float)N_NODES;
  float m = ssum[f] * invn;
  float v = ssq[f] * invn - m * m;
  mean[f] = m;
  istd[f] = rsqrtf(fmaxf(v, 0.0f) + 1e-5f);
}

__global__ void k_bn(float* __restrict__ h, const float* __restrict__ mean,
                     const float* __restrict__ istd, const float* __restrict__ gam,
                     const float* __restrict__ bet, int F, unsigned total) {
  unsigned i = blockIdx.x * 256u + threadIdx.x;
  if (i >= total) return;
  int f = (int)(i % (unsigned)F);
  h[i] = (h[i] - mean[f]) * istd[f] * gam[f] + bet[f];
}

// ---------------------------------------------------------------- layer 7 + pool
__global__ void k_gemv256(const float* __restrict__ h, const float* __restrict__ W,
                          float* __restrict__ out) {
  int i = blockIdx.x * 256 + threadIdx.x;
  if (i >= N_NODES) return;
  const float* row = h + (size_t)i * 256;
  float s = 0.0f;
#pragma unroll 8
  for (int k = 0; k < 256; ++k) s += row[k] * W[k];
  out[i] = s;
}

__global__ void k_scatter1(const float* __restrict__ t, float* __restrict__ agg,
                           const int* __restrict__ src, const int* __restrict__ dst,
                           const float* __restrict__ nrm) {
  unsigned e = blockIdx.x * 256u + threadIdx.x;
  if (e >= N_TOT) return;
  int s, d;
  if (e < N_EDGES) { s = src[e]; d = dst[e]; }
  else             { s = d = (int)(e - N_EDGES); }
  atomicAdd(&agg[d], t[s] * nrm[e]);
}

__global__ void k_zero_out(float* __restrict__ out) {
  if (threadIdx.x < N_GRAPHS) out[threadIdx.x] = 0.0f;
}

__global__ void k_pool(const float* __restrict__ agg, const int* __restrict__ batch,
                       const float* __restrict__ b7, float* __restrict__ out) {
  int i = blockIdx.x * 256 + threadIdx.x;
  if (i < N_NODES) atomicAdd(&out[batch[i]], agg[i] + b7[0]);
}

// ---------------------------------------------------------------- launch
extern "C" void kernel_launch(void* const* d_in, const int* in_sizes, int n_in,
                              void* d_out, int out_size, void* d_ws, size_t ws_size,
                              hipStream_t stream) {
  (void)in_sizes; (void)n_in; (void)out_size; (void)ws_size;

  const float* x     = (const float*)d_in[0];
  const int*   ei    = (const int*)d_in[1];            // [2, E]: src then dst
  const int*   batch = (const int*)d_in[2];
  const float* W[7]; const float* bb[7];
  for (int i = 0; i < 7; ++i) { W[i] = (const float*)d_in[3 + 2 * i]; bb[i] = (const float*)d_in[4 + 2 * i]; }
  const float* gam[6]; const float* bet[6];
  for (int i = 0; i < 6; ++i) { gam[i] = (const float*)d_in[17 + 2 * i]; bet[i] = (const float*)d_in[18 + 2 * i]; }
  float* out = (float*)d_out;

  const int* src = ei;
  const int* dst = ei + N_EDGES;

  // workspace carve-out
  char* ws = (char*)d_ws;
  size_t off = 0;
  auto take = [&](size_t bytes) -> char* {
    char* p = ws + off;
    off = (off + bytes + 255) & ~(size_t)255;
    return p;
  };
  float* bufA  = (float*)take((size_t)N_NODES * 4096 * sizeof(float));   // GEMM output
  float* bufB  = (float*)take((size_t)N_NODES * 4096 * sizeof(float));   // aggregate / layer output
  bf16*  act16 = (bf16*) take((size_t)N_NODES * 4096 * sizeof(bf16));
  bf16*  wt16  = (bf16*) take((size_t)4096 * 2048 * sizeof(bf16));       // transposed weights
  float* nrm   = (float*)take((size_t)N_TOT * sizeof(float));
  float* deg   = (float*)take((size_t)N_NODES * sizeof(float));
  float* dinv  = (float*)take((size_t)N_NODES * sizeof(float));
  float* ssum  = (float*)take((size_t)4096 * sizeof(float));
  float* ssq   = (float*)take((size_t)4096 * sizeof(float));
  float* mean  = (float*)take((size_t)4096 * sizeof(float));
  float* istd  = (float*)take((size_t)4096 * sizeof(float));

  // ---- GCN symmetric normalization
  k_init_deg <<<N_NODES / 256, 256, 0, stream>>>(deg);
  k_count_deg<<<N_EDGES / 256, 256, 0, stream>>>(dst, deg);
  k_dinv     <<<N_NODES / 256, 256, 0, stream>>>(deg, dinv);
  k_norm     <<<(N_TOT + 255) / 256, 256, 0, stream>>>(src, dst, dinv, nrm);

  const int dims[8] = {396, 4096, 2048, 1024, 1024, 2048, 256, 1};
  const float* hin = x;

  for (int L = 0; L < 6; ++L) {
    const int K  = dims[L];
    const int F  = dims[L + 1];
    const int Kp = (K + 31) & ~31;

    unsigned na = (unsigned)N_NODES * (unsigned)Kp;
    k_act_cast<<<(na + 255) / 256, 256, 0, stream>>>(hin, act16, K, Kp);

    dim3 gt(F / 32, Kp / 32);
    dim3 bt(32, 8);
    k_wt_cast_t<<<gt, bt, 0, stream>>>(W[L], wt16, K, Kp, F);

    dim3 gg(F / 64, N_NODES / 64);
    k_gemm<<<gg, 256, 0, stream>>>(act16, wt16, bufA, F, Kp);

    unsigned nt = (unsigned)N_NODES * (unsigned)F;
    k_zero<<<(nt + 255) / 256, 256, 0, stream>>>(bufB, nt);
    k_zero<<<(F + 255) / 256, 256, 0, stream>>>(ssum, (unsigned)F);
    k_zero<<<(F + 255) / 256, 256, 0, stream>>>(ssq, (unsigned)F);

    dim3 gs(N_TOT, F / 256);
    k_scatter<<<gs, 256, 0, stream>>>(bufA, bufB, src, dst, nrm, F);

    dim3 gb(F / 256, N_NODES / 256);
    k_bias_relu_stats<<<gb, 256, 0, stream>>>(bufB, bb[L], ssum, ssq, F);
    k_finalize<<<(F + 255) / 256, 256, 0, stream>>>(ssum, ssq, mean, istd, F);
    k_bn<<<(nt + 255) / 256, 256, 0, stream>>>(bufB, mean, istd, gam[L], bet[L], F, nt);

    hin = bufB;
  }

  // ---- layer 7 (F_out = 1), then global add pool
  k_gemv256<<<N_NODES / 256, 256, 0, stream>>>(hin, W[6], bufA);
  k_zero<<<(N_NODES + 255) / 256, 256, 0, stream>>>(bufB, (unsigned)N_NODES);
  k_scatter1<<<(N_TOT + 255) / 256, 256, 0, stream>>>(bufA, bufB, src, dst, nrm);
  k_zero_out<<<1, 64, 0, stream>>>(out);
  k_pool<<<N_NODES / 256, 256, 0, stream>>>(bufB, batch, bb[6], out);
}
